// GATBackbone_8770323219000
// MI455X (gfx1250) — compile-verified, matrix operand order checked by reading, not run on hardware
//
#include <hip/hip_runtime.h>
#include <hip/hip_bf16.h>

typedef __attribute__((ext_vector_type(16))) __bf16 v16bf;
typedef __attribute__((ext_vector_type(8)))  __bf16 v8bf;
typedef __attribute__((ext_vector_type(8)))  float  v8f;

#define EPSV 1e-5f

__device__ __forceinline__ float wave_bsum(float v) {
#pragma unroll
  for (int m = 16; m >= 1; m >>= 1) v += __shfl_xor(v, m, 32);
  return v;
}

// float atomic max via sign-aware integer atomics (monotone bit patterns)
__device__ __forceinline__ void atomicMaxF(float* addr, float val) {
  if (val >= 0.0f) atomicMax((int*)addr, __float_as_int(val));
  else             atomicMin((unsigned int*)addr, __float_as_uint(val));
}

// ---------------- conversion / weight swizzle ----------------

__global__ void k_f32_to_bf16(const float* __restrict__ in, __bf16* __restrict__ out, int n) {
  int i = blockIdx.x * blockDim.x + threadIdx.x;
  if (i < n) out[i] = (__bf16)in[i];
}

// Pre-swizzle a 128x128 row-major (K x Ncol) weight matrix into the exact
// per-lane WMMA B-operand layout (ISA 7.12.2, 16-bit B 32x16):
//   chunk index = ((coltile*4 + kblock)*32 + lane)*16 + j
//   lane: n = coltile*16 + (lane&15); khalf = (lane>=16)?8:0
//   j<8 -> K = kblock*32 + khalf + j ; j>=8 -> K = kblock*32 + khalf + 8 + j
__global__ void k_swizzle_w(const float* __restrict__ W, __bf16* __restrict__ out) {
  int i = blockIdx.x * blockDim.x + threadIdx.x;
  if (i >= 128 * 128) return;
  int j    = i & 15;
  int lane = (i >> 4) & 31;
  int kb   = (i >> 9) & 3;
  int t    = i >> 11;
  int n = t * 16 + (lane & 15);
  int k = kb * 32 + ((lane >= 16) ? 8 : 0) + ((j < 8) ? j : 8 + j);
  out[i] = (__bf16)W[k * 128 + n];
}

// ---------------- dual GEMM: HL = A*Wl + bl, HR = A*Wr + br ----------------
// grid.x = row tiles (16 rows each), 256 threads = 8 waves, wave w -> col tile w

__global__ void k_gemm_dual_wmma(const __bf16* __restrict__ A,     // [N,128] bf16 row-major
                                 const __bf16* __restrict__ BLsw,  // swizzled 128x128
                                 const __bf16* __restrict__ BRsw,
                                 const float* __restrict__ bl,
                                 const float* __restrict__ br,
                                 float* __restrict__ HL,
                                 float* __restrict__ HR,
                                 int N) {
  int wave = threadIdx.x >> 5;   // col tile 0..7
  int lane = threadIdx.x & 31;
  int rt   = blockIdx.x;
  int row  = rt * 16 + (lane & 15);
  int rowc = (row < N) ? row : (N - 1);
  int khalf = (lane >= 16) ? 8 : 0;

  v8f cl = {};
  v8f cr = {};
#pragma unroll
  for (int kb = 0; kb < 4; ++kb) {
    int kbase = kb * 32 + khalf;
    // A fragment: 16-bit A 16x32 layout -> two contiguous 16B loads per lane
    v8bf a0 = *(const v8bf*)(A + (size_t)rowc * 128 + kbase);
    v8bf a1 = *(const v8bf*)(A + (size_t)rowc * 128 + kbase + 16);
    v16bf a;
#pragma unroll
    for (int q = 0; q < 8; ++q) { a[q] = a0[q]; a[8 + q] = a1[q]; }
    // B fragments: pre-swizzled, 32 contiguous bytes per lane
    v16bf bfl = *(const v16bf*)(BLsw + (((wave * 4 + kb) * 32 + lane) << 4));
    v16bf bfr = *(const v16bf*)(BRsw + (((wave * 4 + kb) * 32 + lane) << 4));
    cl = __builtin_amdgcn_wmma_f32_16x16x32_bf16(false, a, false, bfl, (short)0, cl, false, false);
    cr = __builtin_amdgcn_wmma_f32_16x16x32_bf16(false, a, false, bfr, (short)0, cr, false, false);
  }
  int n   = wave * 16 + (lane & 15);
  float blv = bl[n];
  float brv = br[n];
  int m0 = rt * 16 + khalf;   // C/D layout: VGPR v -> row m0+v
#pragma unroll
  for (int v = 0; v < 8; ++v) {
    int r = m0 + v;
    if (r < N) {
      HL[(size_t)r * 128 + n] = cl[v] + blv;
      HR[(size_t)r * 128 + n] = cr[v] + brv;
    }
  }
}

// ---------------- softmax / output init ----------------

__global__ void k_init_softmax(float* __restrict__ mx, float* __restrict__ s, int n) {
  int i = blockIdx.x * blockDim.x + threadIdx.x;
  if (i < n) { mx[i] = -3.0e38f; s[i] = 0.0f; }
}

__global__ void k_init_out_bias(float* __restrict__ out, const float* __restrict__ bo, int total) {
  int i = blockIdx.x * blockDim.x + threadIdx.x;
  if (i < total) out[i] = bo[i & 127];
}

// ---------------- edge pass 1: logits + scatter-max ----------------
// one wave per edge; lane l covers elements l, l+32, l+64, l+96

__global__ void k_edge_logits(const float* __restrict__ HL, const float* __restrict__ HR,
                              const float* __restrict__ att,   // flat [128]
                              const int* __restrict__ ei, int E, int Et,
                              float* __restrict__ logits, float* __restrict__ mx,
                              int Hh) {
  int e = blockIdx.x * 8 + (threadIdx.x >> 5);
  if (e >= Et) return;
  int lane = threadIdx.x & 31;
  int s, d;
  if (e < E) { s = ei[e]; d = ei[E + e]; } else { s = e - E; d = s; }
  float p[4];
#pragma unroll
  for (int j = 0; j < 4; ++j) {
    int idx = lane + 32 * j;
    float m = HL[(size_t)s * 128 + idx] + HR[(size_t)d * 128 + idx];
    m = (m > 0.0f) ? m : 0.2f * m;                 // leaky_relu(0.2)
    p[j] = m * att[idx];
  }
  if (Hh == 4) {
#pragma unroll
    for (int j = 0; j < 4; ++j) p[j] = wave_bsum(p[j]);
    if (lane < 4) {
      logits[(size_t)e * 4 + lane] = p[lane];
      atomicMaxF(&mx[(size_t)d * 4 + lane], p[lane]);
    }
  } else {
    float t = wave_bsum(p[0] + p[1] + p[2] + p[3]);
    if (lane == 0) {
      logits[e] = t;
      atomicMaxF(&mx[d], t);
    }
  }
}

// ---------------- edge pass 2: exp + scatter-sum ----------------

__global__ void k_edge_exp(const float* __restrict__ logits, const float* __restrict__ mx,
                           const int* __restrict__ ei, int E, int Et,
                           float* __restrict__ expv, float* __restrict__ ssum, int Hh) {
  int i = blockIdx.x * blockDim.x + threadIdx.x;
  if (i >= Et * Hh) return;
  int e = i / Hh;
  int h = i - e * Hh;
  int d = (e < E) ? ei[E + e] : (e - E);
  float ex = __expf(logits[i] - mx[(size_t)d * Hh + h]);
  expv[i] = ex;
  atomicAdd(&ssum[(size_t)d * Hh + h], ex);
}

// ---------------- edge pass 3: alpha-weighted scatter-add ----------------

__global__ void k_edge_scatter(const float* __restrict__ HL,
                               const float* __restrict__ expv, const float* __restrict__ ssum,
                               const int* __restrict__ ei, int E, int Et,
                               float* __restrict__ out, int Hh) {
  int e = blockIdx.x * 8 + (threadIdx.x >> 5);
  if (e >= Et) return;
  int lane = threadIdx.x & 31;
  int s, d;
  if (e < E) { s = ei[e]; d = ei[E + e]; } else { s = e - E; d = s; }
#pragma unroll
  for (int j = 0; j < 4; ++j) {
    int h = (Hh == 4) ? j : 0;
    float alpha = expv[(size_t)e * Hh + h] / (ssum[(size_t)d * Hh + h] + 1e-16f);
    int idx = lane + 32 * j;
    atomicAdd(&out[(size_t)d * 128 + idx], HL[(size_t)s * 128 + idx] * alpha);
  }
}

// ---------------- post: nan_to_num -> layernorm -> relu -> +residual ----------------
// one wave per node (128 elems = 4 per lane)

__global__ void k_post_ln(const float* __restrict__ hraw,
                          const float* __restrict__ g, const float* __restrict__ b,
                          const float* __restrict__ xres,
                          float* __restrict__ xout, int N) {
  int n = blockIdx.x * 8 + (threadIdx.x >> 5);
  if (n >= N) return;
  int lane = threadIdx.x & 31;
  float v[4];
  float sum = 0.0f;
#pragma unroll
  for (int j = 0; j < 4; ++j) {
    float t = hraw[(size_t)n * 128 + lane + 32 * j];
    if (t != t)               t = 0.0f;    // nan
    else if (__builtin_isinf(t)) t = (t > 0.0f) ? EPSV : -EPSV;
    v[j] = t;
    sum += t;
  }
  float mu = wave_bsum(sum) * (1.0f / 128.0f);
  float vs = 0.0f;
#pragma unroll
  for (int j = 0; j < 4; ++j) { float dlt = v[j] - mu; vs += dlt * dlt; }
  vs = wave_bsum(vs) * (1.0f / 128.0f);
  float rs = rsqrtf(vs + EPSV);
#pragma unroll
  for (int j = 0; j < 4; ++j) {
    int idx = lane + 32 * j;
    float y = (v[j] - mu) * rs * g[idx] + b[idx];
    y = fmaxf(y, 0.0f);
    xout[(size_t)n * 128 + idx] = y + xres[(size_t)n * 128 + idx];
  }
}

// ---------------- host driver ----------------

static inline size_t alignup256(size_t x) { return (x + 255) & ~(size_t)255; }

extern "C" void kernel_launch(void* const* d_in, const int* in_sizes, int n_in,
                              void* d_out, int out_size, void* d_ws, size_t ws_size,
                              hipStream_t stream) {
  const float* x  = (const float*)d_in[0];
  const int*   ei = (const int*)d_in[1];
  const int N  = in_sizes[0] / 128;
  const int E  = in_sizes[1] / 2;
  const int Et = E + N;

  // workspace carve-out
  char* p = (char*)d_ws;
  auto take = [&](size_t bytes) -> void* { void* r = (void*)p; p += alignup256(bytes); return r; };
  __bf16* actbf  = (__bf16*)take((size_t)N * 128 * sizeof(__bf16));
  __bf16* wlsw   = (__bf16*)take(128 * 128 * sizeof(__bf16));
  __bf16* wrsw   = (__bf16*)take(128 * 128 * sizeof(__bf16));
  float*  HL     = (float*)take((size_t)N * 128 * sizeof(float));
  float*  HR     = (float*)take((size_t)N * 128 * sizeof(float));
  float*  logits = (float*)take((size_t)Et * 4 * sizeof(float));
  float*  expv   = (float*)take((size_t)Et * 4 * sizeof(float));
  float*  mx     = (float*)take((size_t)N * 4 * sizeof(float));
  float*  ssum   = (float*)take((size_t)N * 4 * sizeof(float));
  float*  x1     = (float*)take((size_t)N * 128 * sizeof(float));
  float*  x2     = (float*)take((size_t)N * 128 * sizeof(float));
  float*  htmp   = (float*)take((size_t)N * 128 * sizeof(float));

  const int T = 256;
  auto run_layer = [&](const float* xin, const float* Wl, const float* bl,
                       const float* Wr, const float* br, const float* att,
                       const float* bo, int Hh, float* outbuf) {
    k_f32_to_bf16<<<(N * 128 + T - 1) / T, T, 0, stream>>>(xin, actbf, N * 128);
    k_swizzle_w  <<<(128 * 128 + T - 1) / T, T, 0, stream>>>(Wl, wlsw);
    k_swizzle_w  <<<(128 * 128 + T - 1) / T, T, 0, stream>>>(Wr, wrsw);
    k_gemm_dual_wmma<<<(N + 15) / 16, T, 0, stream>>>(actbf, wlsw, wrsw, bl, br, HL, HR, N);
    k_init_softmax <<<(N * Hh + T - 1) / T, T, 0, stream>>>(mx, ssum, N * Hh);
    k_init_out_bias<<<(N * 128 + T - 1) / T, T, 0, stream>>>(outbuf, bo, N * 128);
    k_edge_logits <<<(Et + 7) / 8, T, 0, stream>>>(HL, HR, att, ei, E, Et, logits, mx, Hh);
    k_edge_exp    <<<(Et * Hh + T - 1) / T, T, 0, stream>>>(logits, mx, ei, E, Et, expv, ssum, Hh);
    k_edge_scatter<<<(Et + 7) / 8, T, 0, stream>>>(HL, expv, ssum, ei, E, Et, outbuf, Hh);
  };

  // layer 0
  run_layer(x,
            (const float*)d_in[2],  (const float*)d_in[3],
            (const float*)d_in[4],  (const float*)d_in[5],
            (const float*)d_in[6],  (const float*)d_in[7], 4, htmp);
  k_post_ln<<<(N + 7) / 8, T, 0, stream>>>(htmp, (const float*)d_in[8], (const float*)d_in[9], x, x1, N);

  // layer 1
  run_layer(x1,
            (const float*)d_in[10], (const float*)d_in[11],
            (const float*)d_in[12], (const float*)d_in[13],
            (const float*)d_in[14], (const float*)d_in[15], 4, htmp);
  k_post_ln<<<(N + 7) / 8, T, 0, stream>>>(htmp, (const float*)d_in[16], (const float*)d_in[17], x1, x2, N);

  // final layer (heads=1, concat=False) straight into d_out
  run_layer(x2,
            (const float*)d_in[18], (const float*)d_in[19],
            (const float*)d_in[20], (const float*)d_in[21],
            (const float*)d_in[22], (const float*)d_in[23], 1, (float*)d_out);
  (void)n_in; (void)out_size; (void)ws_size;
}
